// ECAPA2AttentiveStatPoolingBlock_6279242187067
// MI455X (gfx1250) — compile-verified, matrix operand order checked by reading, not run on hardware
//
#include <hip/hip_runtime.h>
#include <hip/hip_bf16.h>
#include <math.h>

#define EPSF 1e-5f

#define B_   32
#define C_   1536
#define A_   128
#define T_   1000
#define TP_  1008    // padded T for h layout [B][TP][A]
#define TT_  63      // ceil(1000/16)
#define CT_  96      // 1536/16
#define KC_  48      // 1536/32 k-steps for GEMM1
#define KA_  4       // 128/32  k-steps for GEMM2
#define LDSS 1544    // padded bf16 row stride for GEMM1 x tile
#define LGS  1008    // padded f32 row stride for logits tile

typedef __attribute__((ext_vector_type(16))) __bf16 bfx16;
typedef __attribute__((ext_vector_type(8)))  float  fx8;

union BFrag { bfx16 v; __bf16 e[16]; };

__device__ inline __bf16 to_bf(float f) { return (__bf16)f; }

// ---------------- Kernel 1: per-(b,c) mean / unbiased std over T ----------------
__global__ __launch_bounds__(256) void stats_kernel(const float* __restrict__ x,
                                                    float* __restrict__ mean,
                                                    float* __restrict__ stdv) {
    int wave = threadIdx.x >> 5;
    int lane = threadIdx.x & 31;
    int row = blockIdx.x * 8 + wave;           // row = b*C + c
    if (row >= B_ * C_) return;
    const float* xr = x + (size_t)row * T_;
    float s = 0.f, ss = 0.f;
    for (int t = lane; t < T_; t += 32) {
        float v = xr[t];
        s += v; ss += v * v;
    }
    for (int o = 16; o > 0; o >>= 1) {
        s  += __shfl_down(s,  o, 32);
        ss += __shfl_down(ss, o, 32);
    }
    if (lane == 0) {
        float m = s / (float)T_;
        float var = (ss - (float)T_ * m * m) / (float)(T_ - 1);
        mean[row] = m;
        stdv[row] = sqrtf(var + EPSF);
    }
}

// ---------------- Kernel 2: convert w1(x-part) and w2 to bf16 ----------------
__global__ __launch_bounds__(256) void wconv_kernel(const float* __restrict__ w1,
                                                    const float* __restrict__ w2,
                                                    __bf16* __restrict__ w1xb,
                                                    __bf16* __restrict__ w2b) {
    int i = blockIdx.x * 256 + threadIdx.x;
    if (i < A_ * C_) {
        int a = i / C_, c = i % C_;
        w1xb[i] = to_bf(w1[a * (3 * C_) + c]);
        w2b[i]  = to_bf(w2[i]);              // w2 is [C, A], C*A == A*C
    }
}

// ---------------- Kernel 3: bias[b,a] = b1[a] + w1_mean.mean + w1_std.std ----------------
__global__ __launch_bounds__(256) void biasfold_kernel(const float* __restrict__ w1,
                                                       const float* __restrict__ b1,
                                                       const float* __restrict__ mean,
                                                       const float* __restrict__ stdv,
                                                       float* __restrict__ biasba) {
    int b = blockIdx.x / A_;
    int a = blockIdx.x % A_;
    int tid = threadIdx.x;
    const float* w1r = w1 + (size_t)a * (3 * C_);
    const float* mr = mean + (size_t)b * C_;
    const float* sr = stdv + (size_t)b * C_;
    float p = 0.f;
    for (int c = tid; c < C_; c += 256)
        p += w1r[C_ + c] * mr[c] + w1r[2 * C_ + c] * sr[c];
    __shared__ float red[8];
    int lane = tid & 31, wave = tid >> 5;
    for (int o = 16; o > 0; o >>= 1) p += __shfl_down(p, o, 32);
    if (lane == 0) red[wave] = p;
    __syncthreads();
    if (tid == 0) {
        float t = 0.f;
        for (int i = 0; i < 8; ++i) t += red[i];
        biasba[b * A_ + a] = t + b1[a];
    }
}

// ---------------- Kernel 4: GEMM1 (w1x @ x) + ReLU + BN1 -> h bf16 [B][TP][A] ----------------
// block = (b, t_tile); 8 waves, wave w owns a-tile w. x tile staged bf16 in LDS.
__global__ __launch_bounds__(256) void gemm1_kernel(const float* __restrict__ x,
                                                    const __bf16* __restrict__ w1xb,
                                                    const float* __restrict__ biasba,
                                                    const float* __restrict__ g1,
                                                    const float* __restrict__ be1,
                                                    const float* __restrict__ rm1,
                                                    const float* __restrict__ rv1,
                                                    __bf16* __restrict__ hbf) {
    __shared__ __bf16 xs[16 * LDSS];           // [t_local][c] padded
    int b  = blockIdx.x / TT_;
    int tt = blockIdx.x % TT_;
    int tid = threadIdx.x;
    const float* xb = x + (size_t)b * C_ * T_;

    // stage x[b, :, tt*16 .. +15] -> LDS as bf16 (coalesced over t)
    for (int i = tid; i < C_ * 16; i += 256) {
        int c = i >> 4;
        int tl = i & 15;
        int t = tt * 16 + tl;
        float v = (t < T_) ? xb[(size_t)c * T_ + t] : 0.f;
        xs[tl * LDSS + c] = to_bf(v);
    }
    __syncthreads();

    int wave = tid >> 5;                        // a-tile
    int lane = tid & 31;
    int lh = lane >> 4;                         // half
    int l16 = lane & 15;
    int arow = wave * 16 + l16;                 // A-matrix row for this lane
    const __bf16* w1r = w1xb + (size_t)arow * C_;

    fx8 acc = {};
    for (int k = 0; k < KC_; ++k) {
        BFrag af, bm;
#pragma unroll
        for (int v = 0; v < 8; ++v) {
            int kb = ((v & 4) ? 16 : 0) + 2 * (v & 3) + lh * 8;
            int c = k * 32 + kb;
            af.e[2 * v]     = w1r[c];
            af.e[2 * v + 1] = w1r[c + 1];
            bm.e[2 * v]     = xs[l16 * LDSS + c];
            bm.e[2 * v + 1] = xs[l16 * LDSS + c + 1];
        }
        acc = __builtin_amdgcn_wmma_f32_16x16x32_bf16(false, af.v, false, bm.v,
                                                      (short)0, acc, false, false);
    }

    // epilogue: +bias, ReLU, BN1, store bf16 h[b][t][a] -- 8 contiguous bf16 per lane
    int t = tt * 16 + l16;
    if (t < T_) {
        __bf16* hrow = hbf + ((size_t)b * TP_ + t) * A_ + wave * 16 + 8 * lh;
#pragma unroll
        for (int r = 0; r < 8; ++r) {
            int a = wave * 16 + r + 8 * lh;     // C/D: VGPR r -> M = r + 8*half
            float v = acc[r] + biasba[b * A_ + a];
            v = fmaxf(v, 0.f);
            float sc = g1[a] * rsqrtf(rv1[a] + EPSF);
            v = (v - rm1[a]) * sc + be1[a];
            hrow[r] = to_bf(v);
        }
    }
}

// ---------------- Kernel 5: GEMM2 (w2 @ h) -> logits in LDS -> softmax -> pooled stats + BN2 ----------------
// block = (b, c_tile of 16); 8 waves split the 63 t-tiles; logits [16 x 1000] live in LDS only.
__global__ __launch_bounds__(256) void gemm2_pool_kernel(const float* __restrict__ x,
                                                         const __bf16* __restrict__ w2b,
                                                         const __bf16* __restrict__ hbf,
                                                         const float* __restrict__ b2,
                                                         const float* __restrict__ g2,
                                                         const float* __restrict__ be2,
                                                         const float* __restrict__ rm2,
                                                         const float* __restrict__ rv2,
                                                         float* __restrict__ out) {
    __shared__ float lg[16 * LGS];              // logits [row][t]
    int b  = blockIdx.x / CT_;
    int ct = blockIdx.x % CT_;
    int tid = threadIdx.x;
    int wave = tid >> 5;
    int lane = tid & 31;
    int lh = lane >> 4;
    int l16 = lane & 15;

    // preload w2 A-fragments for all 4 k-steps (rows = ct*16 + l16, K = 128)
    const __bf16* w2r = w2b + (size_t)(ct * 16 + l16) * A_;
    BFrag af[KA_];
#pragma unroll
    for (int k = 0; k < KA_; ++k) {
#pragma unroll
        for (int v = 0; v < 8; ++v) {
            int kb = ((v & 4) ? 16 : 0) + 2 * (v & 3) + lh * 8;
            int a = k * 32 + kb;
            af[k].e[2 * v]     = w2r[a];
            af[k].e[2 * v + 1] = w2r[a + 1];
        }
    }

    const __bf16* hb = hbf + (size_t)b * TP_ * A_;
    for (int tile = wave; tile < TT_; tile += 8) {
        int t0 = tile * 16;
        int t = t0 + l16;
        // h[b][t][a]: per-lane fragment = two contiguous 16B chunks (a = k*32+8*lh+0..7, +16..23)
        const __bf16* hr = hb + (size_t)t * A_;
        fx8 acc = {};
#pragma unroll
        for (int k = 0; k < KA_; ++k) {
            BFrag bm;
#pragma unroll
            for (int v = 0; v < 8; ++v) {
                int kb = ((v & 4) ? 16 : 0) + 2 * (v & 3) + lh * 8;
                int a = k * 32 + kb;
                bm.e[2 * v]     = hr[a];
                bm.e[2 * v + 1] = hr[a + 1];
            }
            acc = __builtin_amdgcn_wmma_f32_16x16x32_bf16(false, af[k].v, false, bm.v,
                                                          (short)0, acc, false, false);
        }
        if (t < T_) {
#pragma unroll
            for (int r = 0; r < 8; ++r) {
                int crow = r + 8 * lh;
                lg[crow * LGS + t] = acc[r] + b2[ct * 16 + crow];
            }
        }
    }
    __syncthreads();

    // softmax over T + attentive pooling; wave handles rows {wave, wave+8}
    const float* xb = x + (size_t)b * C_ * T_;
    for (int row = wave; row < 16; row += 8) {
        const float* lr = lg + row * LGS;
        int cg = ct * 16 + row;
        const float* xr = xb + (size_t)cg * T_;
        float m = -3.4e38f;
        for (int t = lane; t < T_; t += 32) m = fmaxf(m, lr[t]);
        for (int o = 16; o > 0; o >>= 1) m = fmaxf(m, __shfl_xor(m, o, 32));
        float s = 0.f, sx = 0.f, sxx = 0.f;
        for (int t = lane; t < T_; t += 32) {
            float e = __expf(lr[t] - m);
            float xv = xr[t];
            s += e; sx += e * xv; sxx += e * xv * xv;
        }
        for (int o = 16; o > 0; o >>= 1) {
            s   += __shfl_xor(s,   o, 32);
            sx  += __shfl_xor(sx,  o, 32);
            sxx += __shfl_xor(sxx, o, 32);
        }
        if (lane == 0) {
            float mu = sx / s;
            float ssq = sxx / s;
            float sigma = sqrtf(fabsf(ssq - mu * mu) + EPSF);
            // final BN over 2C features
            float sc0 = g2[cg] * rsqrtf(rv2[cg] + EPSF);
            out[(size_t)b * (2 * C_) + cg] = (mu - rm2[cg]) * sc0 + be2[cg];
            int cg2 = C_ + cg;
            float sc1 = g2[cg2] * rsqrtf(rv2[cg2] + EPSF);
            out[(size_t)b * (2 * C_) + cg2] = (sigma - rm2[cg2]) * sc1 + be2[cg2];
        }
    }
}

extern "C" void kernel_launch(void* const* d_in, const int* in_sizes, int n_in,
                              void* d_out, int out_size, void* d_ws, size_t ws_size,
                              hipStream_t stream) {
    const float* x   = (const float*)d_in[0];
    const float* w1  = (const float*)d_in[1];
    const float* b1  = (const float*)d_in[2];
    const float* g1  = (const float*)d_in[3];
    const float* be1 = (const float*)d_in[4];
    const float* rm1 = (const float*)d_in[5];
    const float* rv1 = (const float*)d_in[6];
    const float* w2  = (const float*)d_in[7];
    const float* b2  = (const float*)d_in[8];
    const float* g2  = (const float*)d_in[9];
    const float* be2 = (const float*)d_in[10];
    const float* rm2 = (const float*)d_in[11];
    const float* rv2 = (const float*)d_in[12];
    float* out = (float*)d_out;

    char* ws = (char*)d_ws;
    float*  mean   = (float*)(ws);                               // B*C f32
    float*  stdv   = (float*)(ws + 196608);                      // B*C f32
    float*  biasba = (float*)(ws + 393216);                      // B*A f32
    __bf16* w1xb   = (__bf16*)(ws + 409600);                     // A*C bf16
    __bf16* w2b    = (__bf16*)(ws + 802816);                     // C*A bf16
    __bf16* hbf    = (__bf16*)(ws + 1196032);                    // B*TP*A bf16 (~8.26 MB)

    stats_kernel<<<(B_ * C_) / 8, 256, 0, stream>>>(x, mean, stdv);
    wconv_kernel<<<(A_ * C_ + 255) / 256, 256, 0, stream>>>(w1, w2, w1xb, w2b);
    biasfold_kernel<<<B_ * A_, 256, 0, stream>>>(w1, b1, mean, stdv, biasba);
    gemm1_kernel<<<B_ * TT_, 256, 0, stream>>>(x, w1xb, biasba, g1, be1, rm1, rv1, hbf);
    gemm2_pool_kernel<<<B_ * CT_, 256, 0, stream>>>(x, w2b, hbf, b2, g2, be2, rm2, rv2, out);
}
